// MicroNet_51256139710940
// MI455X (gfx1250) — compile-verified, matrix operand order checked by reading, not run on hardware
//
#include <hip/hip_runtime.h>
#include <hip/hip_bf16.h>

typedef __attribute__((ext_vector_type(16))) _Float16 v16h;
typedef __attribute__((ext_vector_type(8)))  float    v8f;
typedef int v4i_ __attribute__((vector_size(16)));

#define LDP 40        // A-tile LDS row stride in halfs (16B-aligned rows, bank spread)
#define BREC 24       // B-tile LDS record stride in halfs (16 data + 8 pad, 48B)
#define SA_HALFS (128 * LDP)   // 5120 halfs per A buffer
#define SB_HALFS (256 * BREC)  // 6144 halfs per B buffer

// ---------------------------------------------------------------------------
// CDNA5 async global->LDS path (guarded; falls back to sync copy)
// ---------------------------------------------------------------------------
#if defined(__has_builtin)
# if __has_builtin(__builtin_amdgcn_global_load_async_to_lds_b128) && \
     __has_builtin(__builtin_amdgcn_s_wait_asynccnt)
#  define USE_ASYNC 1
# endif
#endif
#ifndef USE_ASYNC
# define USE_ASYNC 0
#endif

#define AS1V(p) ((__attribute__((address_space(1))) v4i_*)(p))
#define AS3V(p) ((__attribute__((address_space(3))) v4i_*)(p))

// ---------------------------------------------------------------------------
// WMMA helpers
// ---------------------------------------------------------------------------
__device__ __forceinline__ v8f wmma_f16(v16h a, v16h b, v8f c) {
    return __builtin_amdgcn_wmma_f32_16x16x32_f16(
        /*neg_a=*/false, a, /*neg_b=*/false, b,
        /*c_mod=*/(short)0, c, /*reuse_a=*/false, /*reuse_b=*/false);
}

// A fragment from LDS tile stored [row][k], row stride LDP halfs.
// lane&15 = row (M), lane>>4 = K-half: elems 0..7 -> K=half*8+0..7,
// elems 8..15 -> K=16+half*8+0..7  => two aligned b128 LDS loads.
__device__ __forceinline__ v16h ldsFragA(const _Float16* lds, int rowBase, int lane) {
    const int row  = rowBase + (lane & 15);
    const int half = (lane >> 4) & 1;
    const uint4* p = (const uint4*)(lds + row * LDP + half * 8);
    union { v16h h; uint4 q[2]; } r;
    r.q[0] = p[0];        // K = half*8 .. half*8+7
    r.q[1] = p[2];        // +16 halfs = 32B => K = 16+half*8 ..
    return r.h;
}

// B fragment from LDS tile pre-packed in fragment order:
// record (ntl*32 + lane) holds this lane's 16 halfs contiguously.
__device__ __forceinline__ v16h ldsFragB(const _Float16* lds, int ntl, int lane) {
    const uint4* p = (const uint4*)(lds + (ntl * 32 + lane) * BREC);
    union { v16h h; uint4 q[2]; } r;
    r.q[0] = p[0];
    r.q[1] = p[1];
    return r.h;
}

// Stage one K-step tile (A: 128x32 row-major; B: 8 ntiles packed) into LDS.
__device__ __forceinline__ void stage_tile(
    const _Float16* __restrict__ A, int lda,
    const _Float16* __restrict__ Bp, int NTt,
    int m0, int n0, int kt, _Float16* sA, _Float16* sB, int tid)
{
#pragma unroll
    for (int it = 0; it < 2; ++it) {
        int ch  = tid + it * 256;            // 512 chunks of 8 halfs
        int row = ch >> 2, cg = ch & 3;
        const _Float16* src = A + (size_t)(m0 + row) * lda + kt * 32 + cg * 8;
        _Float16*       dst = sA + row * LDP + cg * 8;
#if USE_ASYNC
        __builtin_amdgcn_global_load_async_to_lds_b128(AS1V(src), AS3V(dst), 0, 0);
#else
        *(uint4*)dst = *(const uint4*)src;
#endif
    }
    const _Float16* Bt = Bp + (((size_t)kt * NTt + (n0 >> 4)) << 9); // 8 ntiles * 512 halfs
#pragma unroll
    for (int it = 0; it < 2; ++it) {
        int ch   = tid + it * 256;           // 512 chunks of 8 halfs
        int rec  = ch >> 1, part = ch & 1;
        const _Float16* src = Bt + rec * 16 + part * 8;
        _Float16*       dst = sB + rec * BREC + part * 8;
#if USE_ASYNC
        __builtin_amdgcn_global_load_async_to_lds_b128(AS1V(src), AS3V(dst), 0, 0);
#else
        *(uint4*)dst = *(const uint4*)src;
#endif
    }
}

// 128x128 block tile, 256 threads = 8 waves as 4(M) x 2(N); wave owns 32x64.
// Double-buffered LDS with async staging; 8 wmma per wave per K-step.
__device__ __forceinline__ void gemm_tile(
    const _Float16* __restrict__ A, int lda,
    const _Float16* __restrict__ Bp, int NTt,
    int m0, int n0, int ktiles,
    _Float16* sA, _Float16* sB, v8f acc[2][4])
{
    const int tid  = threadIdx.x;
    const int lane = tid & 31;
    const int wave = tid >> 5;
    const int wm   = wave >> 1;   // 0..3
    const int wn   = wave & 1;    // 0..1

#pragma unroll
    for (int i = 0; i < 2; ++i)
#pragma unroll
        for (int j = 0; j < 4; ++j)
#pragma unroll
            for (int e = 0; e < 8; ++e) acc[i][j][e] = 0.0f;

    stage_tile(A, lda, Bp, NTt, m0, n0, 0, sA, sB, tid);

    for (int kt = 0; kt < ktiles; ++kt) {
        _Float16* cA = sA + (kt & 1) * SA_HALFS;
        _Float16* cB = sB + (kt & 1) * SB_HALFS;
        if (kt + 1 < ktiles) {
            stage_tile(A, lda, Bp, NTt, m0, n0, kt + 1,
                       sA + ((kt + 1) & 1) * SA_HALFS,
                       sB + ((kt + 1) & 1) * SB_HALFS, tid);
#if USE_ASYNC
            __builtin_amdgcn_s_wait_asynccnt(4);   // current tile's 4 asyncs done
#endif
        } else {
#if USE_ASYNC
            __builtin_amdgcn_s_wait_asynccnt(0);
#endif
        }
        __syncthreads();

        v16h a0 = ldsFragA(cA, wm * 32,      lane);
        v16h a1 = ldsFragA(cA, wm * 32 + 16, lane);
#pragma unroll
        for (int nt = 0; nt < 4; ++nt) {
            v16h b = ldsFragB(cB, wn * 4 + nt, lane);
            acc[0][nt] = wmma_f16(a0, b, acc[0][nt]);
            acc[1][nt] = wmma_f16(a1, b, acc[1][nt]);
        }
        __syncthreads();
    }
}

__device__ __forceinline__ float gelu_exact(float x) {
    return 0.5f * x * (1.0f + erff(x * 0.70710678118654752f));
}

// ---------------------------------------------------------------------------
// Weight conversion into fragment-ordered packed layout
// Bp[kt][ntile][lane][j]:  n = ntile*16 + (lane&15),
//   k = kt*32 + (j<8 ? (lane>>4)*8 + j : 16 + (lane>>4)*8 + (j-8))
// ---------------------------------------------------------------------------
// fc_w [6625][2048] f32 -> packed f16, N padded to 6656 (NT=416, KT=64)
__global__ void __launch_bounds__(256) cvt_fcw_kernel(const float* __restrict__ fcw,
                                                      _Float16* __restrict__ bp) {
    size_t i = (size_t)blockIdx.x * 256 + threadIdx.x;
    if (i >= (size_t)2048 * 6656) return;
    int j     = (int)(i & 15);
    int lane  = (int)((i >> 4) & 31);
    int ntile = (int)((i >> 9) % 416);
    int kt    = (int)(i / ((size_t)416 * 512));
    int n     = ntile * 16 + (lane & 15);
    int half  = lane >> 4;
    int k     = kt * 32 + (j < 8 ? half * 8 + j : 16 + half * 8 + (j - 8));
    bp[i] = (n < 6625) ? (_Float16)fcw[(size_t)n * 2048 + k] : (_Float16)0.0f;
}

// pw_w [4][co=256][ci=256] f32 -> packed f16 per block (NT=16, KT=8)
__global__ void __launch_bounds__(256) cvt_pww_kernel(const float* __restrict__ pw,
                                                      _Float16* __restrict__ bp) {
    int i = blockIdx.x * 256 + threadIdx.x;        // 4*65536 total
    int j     = i & 15;
    int lane  = (i >> 4) & 31;
    int ntile = (i >> 9) & 15;
    int kt    = (i >> 13) & 7;
    int blk   = i >> 16;
    int n     = ntile * 16 + (lane & 15);
    int half  = lane >> 4;
    int k     = kt * 32 + (j < 8 ? half * 8 + j : 16 + half * 8 + (j - 8));
    bp[i] = (_Float16)pw[blk * 65536 + n * 256 + k];
}

// ---------------------------------------------------------------------------
// Stem: 4x4 stride-4 conv + bias + BN + GELU, output channels-last f16
// ---------------------------------------------------------------------------
__global__ void __launch_bounds__(256) stem_kernel(
    const float* __restrict__ x, const float* __restrict__ w,
    const float* __restrict__ bias,
    const float* __restrict__ g, const float* __restrict__ be,
    const float* __restrict__ mm, const float* __restrict__ vv,
    _Float16* __restrict__ y)
{
    __shared__ float patch[48];
    const int pos = blockIdx.x;
    const int b  = pos >> 11;
    const int rem = pos & 2047;
    const int oh = rem >> 8, ow = rem & 255;
    const int tid = threadIdx.x;
    if (tid < 48) {
        int ic = tid >> 4, kh = (tid >> 2) & 3, kw = tid & 3;
        patch[tid] = x[(((size_t)b * 3 + ic) * 32 + oh * 4 + kh) * 1024 + ow * 4 + kw];
    }
    __syncthreads();
    const int c = tid;
    float acc = bias[c];
    const float* wc = w + c * 48;
#pragma unroll
    for (int j = 0; j < 48; ++j) acc += patch[j] * wc[j];
    float sc = g[c] * rsqrtf(vv[c] + 1e-5f);
    acc = (acc - mm[c]) * sc + be[c];
    y[(size_t)pos * 256 + c] = (_Float16)gelu_exact(acc);
}

// ---------------------------------------------------------------------------
// Depthwise 3x3 (pad 1) + bias + BN + GELU + residual, channels-last f16
// ---------------------------------------------------------------------------
__global__ void __launch_bounds__(256) dw_kernel(
    const _Float16* __restrict__ yin, _Float16* __restrict__ yout,
    const float* __restrict__ w, const float* __restrict__ bias,
    const float* __restrict__ g, const float* __restrict__ be,
    const float* __restrict__ mm, const float* __restrict__ vv)
{
    const int idx = blockIdx.x * 256 + threadIdx.x;
    const int c = idx & 255;
    const int p = idx >> 8;
    const int b = p >> 11;
    const int rem = p & 2047;
    const int h = rem >> 8, wd = rem & 255;
    float acc = bias[c];
#pragma unroll
    for (int dh = -1; dh <= 1; ++dh) {
        int hh = h + dh;
        if (hh < 0 || hh >= 8) continue;
#pragma unroll
        for (int dw2 = -1; dw2 <= 1; ++dw2) {
            int ww = wd + dw2;
            if (ww < 0 || ww >= 256) continue;
            float xin = (float)yin[(((size_t)b * 8 + hh) * 256 + ww) * 256 + c];
            acc += xin * w[c * 9 + (dh + 1) * 3 + (dw2 + 1)];
        }
    }
    float sc = g[c] * rsqrtf(vv[c] + 1e-5f);
    acc = (acc - mm[c]) * sc + be[c];
    acc = gelu_exact(acc);
    yout[(size_t)idx] = (_Float16)((float)yin[(size_t)idx] + acc);
}

// ---------------------------------------------------------------------------
// Pointwise 1x1 conv as WMMA GEMM (M=65536, K=N=256) + bias + BN + GELU
// ---------------------------------------------------------------------------
__global__ void __launch_bounds__(256) pw_gemm_kernel(
    const _Float16* __restrict__ A, const _Float16* __restrict__ Bp,
    _Float16* __restrict__ out,
    const float* __restrict__ bias,
    const float* __restrict__ g, const float* __restrict__ be,
    const float* __restrict__ mm, const float* __restrict__ vv)
{
    __shared__ __align__(16) _Float16 sA[2 * SA_HALFS];
    __shared__ __align__(16) _Float16 sB[2 * SB_HALFS];
    v8f acc[2][4];
    const int m0 = blockIdx.x * 128, n0 = blockIdx.y * 128;
    gemm_tile(A, 256, Bp, 16, m0, n0, 8, sA, sB, acc);

    const int lane = threadIdx.x & 31;
    const int wave = threadIdx.x >> 5;
    const int wm = wave >> 1, wn = wave & 1;
    const int half = lane >> 4;
#pragma unroll
    for (int mt = 0; mt < 2; ++mt)
#pragma unroll
        for (int nt = 0; nt < 4; ++nt) {
            int n = n0 + wn * 64 + nt * 16 + (lane & 15);
            float sc = g[n] * rsqrtf(vv[n] + 1e-5f);
            float sh = be[n] - mm[n] * sc;
            float bz = bias[n];
#pragma unroll
            for (int r = 0; r < 8; ++r) {
                int m = m0 + wm * 32 + mt * 16 + half * 8 + r;
                float xv = (acc[mt][nt][r] + bz) * sc + sh;
                out[(size_t)m * 256 + n] = (_Float16)gelu_exact(xv);
            }
        }
}

// ---------------------------------------------------------------------------
// ConvLSTM over 256 channel "time" steps; one thread per (b,l).
// ---------------------------------------------------------------------------
__global__ void __launch_bounds__(256) lstm_kernel(
    const _Float16* __restrict__ y, const float* __restrict__ Wl,
    const float* __restrict__ bl, _Float16* __restrict__ feat)
{
    __shared__ float sW[512];
    __shared__ float sb[32];
    const int tid = threadIdx.x;
    for (int j = tid; j < 512; j += 256) sW[j] = Wl[j];
    if (tid < 32) sb[tid] = bl[tid];
    __syncthreads();

    const int gpos = blockIdx.x * 256 + tid;   // b*256 + l
    const int b = gpos >> 8, l = gpos & 255;
    float h[8], cs[8];
#pragma unroll
    for (int f = 0; f < 8; ++f) { h[f] = 0.0f; cs[f] = 0.0f; }

    for (int t = 0; t < 256; ++t) {
        float comb[16];
#pragma unroll
        for (int f = 0; f < 8; ++f)
            comb[f] = (float)y[(((size_t)b * 8 + f) * 256 + l) * 256 + t];
#pragma unroll
        for (int f = 0; f < 8; ++f) comb[8 + f] = h[f];
        float Aa[32];
#pragma unroll
        for (int o = 0; o < 32; ++o) {
            float s = sb[o];
#pragma unroll
            for (int cc = 0; cc < 16; ++cc) s += sW[o * 16 + cc] * comb[cc];
            Aa[o] = s;
        }
#pragma unroll
        for (int f = 0; f < 8; ++f) {
            float ig = 1.0f / (1.0f + expf(-Aa[f]));
            float fg = 1.0f / (1.0f + expf(-Aa[8 + f]));
            float og = 1.0f / (1.0f + expf(-Aa[16 + f]));
            float gg = tanhf(Aa[24 + f]);
            cs[f] = fg * cs[f] + ig * gg;
            h[f]  = og * tanhf(cs[f]);
            feat[(size_t)gpos * 2048 + t * 8 + f] = (_Float16)h[f];
        }
    }
}

// ---------------------------------------------------------------------------
// Final FC: WMMA GEMM M=8192, K=2048, N=6625 (B packed/padded to 6656) + fc_b
// ---------------------------------------------------------------------------
__global__ void __launch_bounds__(256) fc_gemm_kernel(
    const _Float16* __restrict__ A, const _Float16* __restrict__ Bp,
    const float* __restrict__ bias, float* __restrict__ out)
{
    __shared__ __align__(16) _Float16 sA[2 * SA_HALFS];
    __shared__ __align__(16) _Float16 sB[2 * SB_HALFS];
    v8f acc[2][4];
    const int m0 = blockIdx.x * 128, n0 = blockIdx.y * 128;
    gemm_tile(A, 2048, Bp, 416, m0, n0, 64, sA, sB, acc);

    const int lane = threadIdx.x & 31;
    const int wave = threadIdx.x >> 5;
    const int wm = wave >> 1, wn = wave & 1;
    const int half = lane >> 4;
#pragma unroll
    for (int mt = 0; mt < 2; ++mt)
#pragma unroll
        for (int nt = 0; nt < 4; ++nt) {
            int n = n0 + wn * 64 + nt * 16 + (lane & 15);
            if (n >= 6625) continue;
            float bz = bias[n];
#pragma unroll
            for (int r = 0; r < 8; ++r) {
                int m = m0 + wm * 32 + mt * 16 + half * 8 + r;
                out[(size_t)m * 6625 + n] = acc[mt][nt][r] + bz;
            }
        }
}

// ---------------------------------------------------------------------------
// Launcher
// ---------------------------------------------------------------------------
extern "C" void kernel_launch(void* const* d_in, const int* in_sizes, int n_in,
                              void* d_out, int out_size, void* d_ws, size_t ws_size,
                              hipStream_t stream) {
    (void)in_sizes; (void)n_in; (void)out_size; (void)ws_size;
    const float* x       = (const float*)d_in[0];
    const float* conv0_w = (const float*)d_in[1];
    const float* conv0_b = (const float*)d_in[2];
    const float* bn0_g   = (const float*)d_in[3];
    const float* bn0_b   = (const float*)d_in[4];
    const float* bn0_m   = (const float*)d_in[5];
    const float* bn0_v   = (const float*)d_in[6];
    const float* dw_w    = (const float*)d_in[7];
    const float* dw_b    = (const float*)d_in[8];
    const float* bn1_g   = (const float*)d_in[9];
    const float* bn1_b   = (const float*)d_in[10];
    const float* bn1_m   = (const float*)d_in[11];
    const float* bn1_v   = (const float*)d_in[12];
    const float* pw_w    = (const float*)d_in[13];
    const float* pw_b    = (const float*)d_in[14];
    const float* bn2_g   = (const float*)d_in[15];
    const float* bn2_b   = (const float*)d_in[16];
    const float* bn2_m   = (const float*)d_in[17];
    const float* bn2_v   = (const float*)d_in[18];
    const float* lstm_w  = (const float*)d_in[19];
    const float* lstm_b  = (const float*)d_in[20];
    const float* fc_w    = (const float*)d_in[21];
    const float* fc_b    = (const float*)d_in[22];

    char* ws = (char*)d_ws;
    const size_t YB = (size_t)65536 * 256 * sizeof(_Float16);   // 33,554,432 B
    _Float16* y0   = (_Float16*)(ws);
    _Float16* y1   = (_Float16*)(ws + YB);
    _Float16* feat = (_Float16*)(ws + 2 * YB);                  // 8192*2048 f16
    _Float16* fcP  = (_Float16*)(ws + 3 * YB);                  // 2048*6656 f16 packed
    _Float16* pwP  = (_Float16*)(ws + 3 * YB + (size_t)2048 * 6656 * sizeof(_Float16));

    // Weight packing
    cvt_fcw_kernel<<<(2048 * 6656 + 255) / 256, 256, 0, stream>>>(fc_w, fcP);
    cvt_pww_kernel<<<(4 * 65536) / 256, 256, 0, stream>>>(pw_w, pwP);

    // Stem
    stem_kernel<<<65536, 256, 0, stream>>>(x, conv0_w, conv0_b,
                                           bn0_g, bn0_b, bn0_m, bn0_v, y0);
    // MicroBlocks
    for (int i = 0; i < 4; ++i) {
        dw_kernel<<<65536, 256, 0, stream>>>(y0, y1,
            dw_w + i * 256 * 9, dw_b + i * 256,
            bn1_g + i * 256, bn1_b + i * 256, bn1_m + i * 256, bn1_v + i * 256);
        pw_gemm_kernel<<<dim3(512, 2), 256, 0, stream>>>(y1, pwP + i * 65536, y0,
            pw_b + i * 256,
            bn2_g + i * 256, bn2_b + i * 256, bn2_m + i * 256, bn2_v + i * 256);
    }
    // ConvLSTM
    lstm_kernel<<<32, 256, 0, stream>>>(y0, lstm_w, lstm_b, feat);
    // Final FC
    fc_gemm_kernel<<<dim3(64, 52), 256, 0, stream>>>(feat, fcP, fc_b, (float*)d_out);
}
